// RPN_64424509440803
// MI455X (gfx1250) — matrix operand order repaired and checked by reading, not compiled, not run on hardware
//
#include <hip/hip_runtime.h>
#include <hip/hip_bf16.h>

// ---------------------------------------------------------------------------
// RPN head for MI455X (gfx1250, wave32, WMMA).
// conv1 (3x3 512->512) and the fused cls/box 1x1 convs are implicit-im2col
// GEMMs on v_wmma_f32_16x16x32_bf16. Weights pre-swizzled into per-wave
// fragment order with K-step outermost so the hot loop's B loads are
// immediate-offset b128 loads off a single induction pointer.
// ---------------------------------------------------------------------------

typedef __attribute__((ext_vector_type(16))) __bf16 v16bf;
typedef __attribute__((ext_vector_type(8)))  float  v8f;

union Frag16 { v16bf v; unsigned u[8]; };

// ----------------------------- constants -----------------------------------
#define HF        100
#define WF        100
#define NPOS      10000          // HF*WF
#define CIN       512
#define COUT      512
#define KDIM      4608           // CIN*9
#define KS1       144            // KDIM/32 (wmma K-steps, conv1)
#define KS2       16             // CIN/32  (wmma K-steps, gemm2)
#define NANCH     90000          // NPOS*9
#define NSORT     131072         // next pow2 >= NANCH
#define PRE_NMS   12000
#define POST_NMS  2000
#define IMG_SZ    1600.0f
#define PATCH     16.0f          // IMG_SZ / min(HF,WF)
#define IOU_THR   0.7f
#define N2        48             // padded (9 cls + 36 box -> 48)

// ------------------------------ helpers ------------------------------------
__device__ __forceinline__ unsigned short f32_to_bf16_rne(float f) {
  unsigned u = __float_as_uint(f);
  if ((u & 0x7F800000u) == 0x7F800000u) return (unsigned short)(u >> 16); // inf/nan
  unsigned r = u + 0x7FFFu + ((u >> 16) & 1u);
  return (unsigned short)(r >> 16);
}
__device__ __forceinline__ unsigned f32_orderable(float f) {
  unsigned u = __float_as_uint(f);
  return (u & 0x80000000u) ? ~u : (u | 0x80000000u);
}
__device__ __forceinline__ float orderable_f32(unsigned e) {
  unsigned u = (e & 0x80000000u) ? (e ^ 0x80000000u) : ~e;
  return __uint_as_float(u);
}
// fragment slot j (0..15) within a lane -> K offset inside a 32-K step
// (ISA 16-bit A/B layout: lane-half hi owns K runs [hi*8,+8) and [16+hi*8,+8))
__device__ __forceinline__ int frag_kk(int j, int hi) {
  return (j < 8) ? (hi * 8 + j) : (8 + hi * 8 + j);
}

// -------------------------- cast / pack kernels -----------------------------
__global__ void k_cast_bf16(const float* __restrict__ src,
                            unsigned short* __restrict__ dst, int n) {
  int i = blockIdx.x * blockDim.x + threadIdx.x;
  if (i < n) dst[i] = f32_to_bf16_rne(src[i]);
}

// W1 -> fragment-swizzled, K-step outermost: [ks(144)][nt(32)][lane(32)][j(16)]
__global__ void k_pack_w1(const float* __restrict__ w1,
                          unsigned short* __restrict__ W1swz) {
  int i = blockIdx.x * blockDim.x + threadIdx.x;           // < 144*32*32*16
  if (i >= KS1 * (COUT / 16) * 32 * 16) return;
  int j = i & 15, lane = (i >> 4) & 31, t = i >> 9;
  int nt = t & 31, ks = t >> 5;
  int n = nt * 16 + (lane & 15), hi = lane >> 4;
  int k = ks * 32 + frag_kk(j, hi);
  W1swz[i] = f32_to_bf16_rne(w1[(size_t)n * KDIM + k]);
}

// W2 (cls 9 + box 36 + pad 3) -> fragment-swizzled: [nt(3)][ks(16)][lane][j]
__global__ void k_pack_w2(const float* __restrict__ cls_w, const float* __restrict__ box_w,
                          const float* __restrict__ cls_b, const float* __restrict__ box_b,
                          unsigned short* __restrict__ W2swz, float* __restrict__ bias2) {
  int i = blockIdx.x * blockDim.x + threadIdx.x;           // < 3*16*32*16
  if (i < (N2 / 16) * KS2 * 32 * 16) {
    int j = i & 15, lane = (i >> 4) & 31, t = i >> 9;
    int ks = t & 15, nt = t >> 4;
    int n = nt * 16 + (lane & 15), hi = lane >> 4;
    int k = ks * 32 + frag_kk(j, hi);
    float v = (n < 9) ? cls_w[n * CIN + k] : (n < 45 ? box_w[(n - 9) * CIN + k] : 0.0f);
    W2swz[i] = f32_to_bf16_rne(v);
  }
  if (i < N2) bias2[i] = (i < 9) ? cls_b[i] : (i < 45 ? box_b[i - 9] : 0.0f);
}

// ---------------- conv1 as implicit-im2col GEMM (WMMA bf16) -----------------
// M = 10000, N = 512, K = 4608 (k = cin*9 + kh*3 + kw).
// Block: 256 threads = 8 waves; block owns a 16-row M tile; each wave owns 4
// N-tiles. A staged in LDS in fragment order, 64 K per barrier round.
// All 10 fragments of a round are preloaded before the 8 WMMAs so loads
// overlap the matrix pipe; B loads use immediate offsets off one base.
__global__ __launch_bounds__(256) void k_conv1_wmma(
    const unsigned short* __restrict__ Xbf,    // [512][100][100] bf16
    const unsigned short* __restrict__ W1swz,  // [ks][nt][lane][16] bf16
    const float* __restrict__ b1,
    unsigned short* __restrict__ Xswz)         // gemm2-A fragment-swizzled act.
{
  __shared__ __align__(16) unsigned short Asub[2 * 32 * 16];   // 2 K-steps
  const int m0   = blockIdx.x * 16;
  const int tid  = threadIdx.x;
  const int lane = tid & 31;
  const int wid  = tid >> 5;
  const int row  = lane & 15;
  const int hi   = lane >> 4;

  v8f acc[4] = {v8f{}, v8f{}, v8f{}, v8f{}};

  // wave's B base: tiles wid*4..wid*4+3 at byte offsets t*1024; +32768B per
  // K-step; advance 64KB per 64-K round.
  const unsigned* wbase = (const unsigned*)(W1swz + (size_t)wid * 4 * 512 + lane * 16);
  const unsigned* As    = (const unsigned*)(Asub + lane * 16);

  for (int k0 = 0; k0 < KDIM; k0 += 64) {
    // cooperative im2col stage of 2 fragment-ordered A tiles (16x32 each)
    for (int e = tid; e < 1024; e += 256) {
      int sub = e >> 9;                 // which 32-K step
      int ee  = e & 511;
      int sl  = ee >> 4;                // dest lane
      int j   = ee & 15;                // dest slot
      int r   = sl & 15;                // tile row (M)
      int shi = sl >> 4;
      int k   = k0 + sub * 32 + frag_kk(j, shi);
      int cin = k / 9;
      int rem = k - cin * 9;
      int kh  = rem / 3;
      int kw  = rem - kh * 3;
      int p   = m0 + r;
      int h   = p / WF;
      int w   = p - h * WF;
      int hh  = h + kh - 1;
      int ww  = w + kw - 1;
      unsigned short v = 0;
      if (hh >= 0 && hh < HF && ww >= 0 && ww < WF)
        v = Xbf[cin * NPOS + hh * WF + ww];
      Asub[e] = v;
    }
    __syncthreads();

    Frag16 fa0, fa1, fb[8];
#pragma unroll
    for (int v = 0; v < 8; ++v) { fa0.u[v] = As[v]; fa1.u[v] = As[256 + v]; }
#pragma unroll
    for (int q = 0; q < 8; ++q) {
      int sub = q >> 2, t = q & 3;
#pragma unroll
      for (int v = 0; v < 8; ++v)
        fb[q].u[v] = wbase[sub * 8192 + t * 256 + v];   // imm offsets
    }
#pragma unroll
    for (int t = 0; t < 4; ++t)
      acc[t] = __builtin_amdgcn_wmma_f32_16x16x32_bf16(
          false, fa0.v, false, fb[t].v, (short)0, acc[t], false, false);
#pragma unroll
    for (int t = 0; t < 4; ++t)
      acc[t] = __builtin_amdgcn_wmma_f32_16x16x32_bf16(
          false, fa1.v, false, fb[4 + t].v, (short)0, acc[t], false, false);

    __syncthreads();
    wbase += 16384;                                     // 64KB per round
  }

  // epilogue: bias + relu, scatter into gemm2 A-fragment-swizzled layout
  // Xswz index for logical (p, c): mt=p/16, row2=p%16, ks=c>>5, kk=c&31,
  // hi2=(kk>>3)&1, j=(kk&7)|((kk>>4)<<3), lane2=row2+16*hi2.
#pragma unroll
  for (int t = 0; t < 4; ++t) {
    int c = (wid * 4 + t) * 16 + row;
    float bc = b1[c];
    int ks  = c >> 5;
    int kk  = c & 31;
    int hi2 = (kk >> 3) & 1;
    int j   = (kk & 7) | ((kk >> 4) << 3);
#pragma unroll
    for (int r = 0; r < 8; ++r) {
      int row2  = r + hi * 8;
      int lane2 = row2 + 16 * hi2;
      float val = fmaxf(acc[t][r] + bc, 0.0f);
      Xswz[(((size_t)blockIdx.x * KS2 + ks) * 32 + lane2) * 16 + j] = f32_to_bf16_rne(val);
    }
  }
}

// ---------------- fused cls/box 1x1 convs as GEMM (WMMA bf16) ---------------
// M = 10000, N = 48, K = 512. One wave per 16x16 tile; both operands in
// fragment-swizzled order; per-ks stride 1KB -> immediate offsets.
__global__ __launch_bounds__(256) void k_gemm2_wmma(
    const unsigned short* __restrict__ Xswz,   // [625][16][32][16] bf16
    const unsigned short* __restrict__ W2swz,  // [3][16][32][16] bf16
    const float* __restrict__ bias2,
    float* __restrict__ out)                   // d_out
{
  const int NTILES_M = NPOS / 16;              // 625
  const int NTILES_N = N2 / 16;                // 3
  int gw = (blockIdx.x * blockDim.x + threadIdx.x) >> 5;
  if (gw >= NTILES_M * NTILES_N) return;       // wave-uniform
  int mt = gw / NTILES_N, nt = gw - mt * NTILES_N;
  int lane = threadIdx.x & 31;
  int row = lane & 15, hi = lane >> 4;

  const unsigned* Ar = (const unsigned*)(Xswz  + (size_t)mt * KS2 * 512 + lane * 16);
  const unsigned* Br = (const unsigned*)(W2swz + (size_t)nt * KS2 * 512 + lane * 16);

  v8f acc = {};
#pragma unroll 4
  for (int ks = 0; ks < KS2; ++ks) {
    Frag16 fa, fb;
#pragma unroll
    for (int v = 0; v < 8; ++v) { fa.u[v] = Ar[ks * 256 + v]; fb.u[v] = Br[ks * 256 + v]; }
    acc = __builtin_amdgcn_wmma_f32_16x16x32_bf16(
        false, fa.v, false, fb.v, (short)0, acc, false, false);
  }

  int j = nt * 16 + row;
  if (j < 45) {
    float bj = bias2[j];
#pragma unroll
    for (int r = 0; r < 8; ++r) {
      int p = mt * 16 + r + hi * 8;
      float val = acc[r] + bj;
      if (j < 9) {
        out[p * 9 + j] = 1.0f / (1.0f + __expf(-val));          // objectness
      } else {
        int a = j - 9;                                          // anchor*4+coord
        out[NANCH + (p * 9 + (a >> 2)) * 4 + (a & 3)] = val;    // deltas [N][4]
      }
    }
  }
}

// -------------------- decode + clip + validity + sort keys ------------------
__global__ void k_decode(const float* __restrict__ anchors,
                         const float* __restrict__ out,          // d_out (obj+deltas)
                         float* __restrict__ proposals,
                         unsigned long long* __restrict__ keys) {
  int n = blockIdx.x * blockDim.x + threadIdx.x;
  if (n >= NSORT) return;
  if (n >= NANCH) { keys[n] = 0ull; return; }    // pad below everything

  float a0 = anchors[n * 4 + 0], a1 = anchors[n * 4 + 1];
  float a2 = anchors[n * 4 + 2], a3 = anchors[n * 4 + 3];
  const float* d = out + NANCH + n * 4;
  float ah = a2 - a0, aw = a3 - a1;
  float acy = a0 + 0.5f * ah, acx = a1 + 0.5f * aw;
  float cy = d[0] * ah + acy, cx = d[1] * aw + acx;
  float h = __expf(d[2]) * ah, w = __expf(d[3]) * aw;
  float y1 = fminf(fmaxf(cy - 0.5f * h, 0.0f), IMG_SZ);
  float x1 = fminf(fmaxf(cx - 0.5f * w, 0.0f), IMG_SZ);
  float y2 = fminf(fmaxf(cy + 0.5f * h, 0.0f), IMG_SZ);
  float x2 = fminf(fmaxf(cx + 0.5f * w, 0.0f), IMG_SZ);
  proposals[n * 4 + 0] = y1; proposals[n * 4 + 1] = x1;
  proposals[n * 4 + 2] = y2; proposals[n * 4 + 3] = x2;

  bool valid = ((y2 - y1) >= PATCH) && ((x2 - x1) >= PATCH);
  float score = valid ? out[n] : -INFINITY;
  // descending sort, ties -> lower index first (top_k semantics)
  keys[n] = ((unsigned long long)f32_orderable(score) << 32) | (unsigned)(~(unsigned)n);
}

// --------------------------- bitonic sort step ------------------------------
__global__ void k_bitonic(unsigned long long* __restrict__ keys, int k, int j) {
  int i = blockIdx.x * blockDim.x + threadIdx.x;
  int ixj = i ^ j;
  if (ixj > i) {
    unsigned long long a = keys[i], b = keys[ixj];
    bool desc = (i & k) == 0;
    if (desc ? (a < b) : (a > b)) { keys[i] = b; keys[ixj] = a; }
  }
}

// ------------------------- gather top-12000 for NMS -------------------------
__global__ void k_prep_nms(const unsigned long long* __restrict__ keys,
                           const float* __restrict__ proposals,
                           float* __restrict__ boxes, float* __restrict__ live,
                           float* __restrict__ areas) {
  int i = blockIdx.x * blockDim.x + threadIdx.x;
  if (i >= PRE_NMS) return;
  unsigned long long key = keys[i];
  unsigned idx = ~(unsigned)key;
  if (idx >= NANCH) idx = 0;                     // safety; top-12000 are always real
  float y1 = proposals[idx * 4 + 0], x1 = proposals[idx * 4 + 1];
  float y2 = proposals[idx * 4 + 2], x2 = proposals[idx * 4 + 3];
  boxes[i * 4 + 0] = y1; boxes[i * 4 + 1] = x1;
  boxes[i * 4 + 2] = y2; boxes[i * 4 + 3] = x2;
  areas[i] = (y2 - y1) * (x2 - x1);
  live[i]  = orderable_f32((unsigned)(key >> 32));
}

// --------------------------- greedy NMS (serial) ----------------------------
__global__ __launch_bounds__(1024) void k_nms(const float* __restrict__ boxes,
                                              const float* __restrict__ areas,
                                              float* __restrict__ live,
                                              float* __restrict__ post) {
  __shared__ float sv[1024];
  __shared__ int   si[1024];
  __shared__ float sb[6];   // y1,x1,y2,x2,area_i,valid
  int tid = threadIdx.x;

  for (int it = 0; it < POST_NMS; ++it) {
    // argmax(live) with first-index tie-break
    float best = -INFINITY; int bidx = 0x7fffffff;
    for (int n = tid; n < PRE_NMS; n += 1024) {
      float v = live[n];
      if (v > best || (v == best && n < bidx)) { best = v; bidx = n; }
    }
    sv[tid] = best; si[tid] = bidx;
    __syncthreads();
    for (int s = 512; s > 0; s >>= 1) {
      if (tid < s) {
        float v2 = sv[tid + s]; int i2 = si[tid + s];
        if (v2 > sv[tid] || (v2 == sv[tid] && i2 < si[tid])) { sv[tid] = v2; si[tid] = i2; }
      }
      __syncthreads();
    }
    if (tid == 0) {
      int i = si[0];
      sb[0] = boxes[i * 4 + 0]; sb[1] = boxes[i * 4 + 1];
      sb[2] = boxes[i * 4 + 2]; sb[3] = boxes[i * 4 + 3];
      sb[4] = areas[i];
      sb[5] = (sv[0] > -INFINITY) ? 1.0f : 0.0f;
    }
    __syncthreads();

    int i = si[0];
    float y1 = sb[0], x1 = sb[1], y2 = sb[2], x2 = sb[3], ai = sb[4];
    for (int n = tid; n < PRE_NMS; n += 1024) {
      float by1 = boxes[n * 4 + 0], bx1 = boxes[n * 4 + 1];
      float by2 = boxes[n * 4 + 2], bx2 = boxes[n * 4 + 3];
      float inter = fmaxf(fminf(y2, by2) - fmaxf(y1, by1), 0.0f) *
                    fmaxf(fminf(x2, bx2) - fmaxf(x1, bx1), 0.0f);
      float uni = areas[n] + ai - inter;
      float iou = inter / fmaxf(uni, 1e-9f);
      if (iou > IOU_THR || n == i) live[n] = -INFINITY;
    }
    if (tid < 4) post[it * 4 + tid] = sb[5] * sb[tid];
    __syncthreads();
  }
}

// ------------------------------- launcher -----------------------------------
static inline size_t align256(size_t x) { return (x + 255) & ~(size_t)255; }

extern "C" void kernel_launch(void* const* d_in, const int* in_sizes, int n_in,
                              void* d_out, int out_size, void* d_ws, size_t ws_size,
                              hipStream_t stream) {
  (void)in_sizes; (void)n_in; (void)out_size; (void)ws_size;

  const float* feature_map = (const float*)d_in[0];   // [1,512,100,100]
  const float* anchor_map  = (const float*)d_in[1];   // [90000,4]
  const float* conv1_w     = (const float*)d_in[2];   // [512,512,3,3]
  const float* conv1_b     = (const float*)d_in[3];   // [512]
  const float* cls_w       = (const float*)d_in[4];   // [9,512,1,1]
  const float* cls_b       = (const float*)d_in[5];   // [9]
  const float* box_w       = (const float*)d_in[6];   // [36,512,1,1]
  const float* box_b       = (const float*)d_in[7];   // [36]
  float* out = (float*)d_out;                         // 90000 + 360000 + 8000 f32

  // workspace layout
  char* ws = (char*)d_ws;
  size_t off = 0;
  unsigned short* Xbf   = (unsigned short*)(ws + off); off = align256(off + (size_t)CIN * NPOS * 2);
  unsigned short* W1swz = (unsigned short*)(ws + off); off = align256(off + (size_t)COUT * KDIM * 2);
  unsigned short* W2swz = (unsigned short*)(ws + off); off = align256(off + (size_t)N2 * CIN * 2);
  float* bias2          = (float*)(ws + off);          off = align256(off + (size_t)N2 * 4);
  unsigned short* Xswz  = (unsigned short*)(ws + off); off = align256(off + (size_t)NPOS * CIN * 2);
  float* proposals      = (float*)(ws + off);          off = align256(off + (size_t)NANCH * 4 * 4);
  unsigned long long* keys = (unsigned long long*)(ws + off); off = align256(off + (size_t)NSORT * 8);
  float* pre_boxes      = (float*)(ws + off);          off = align256(off + (size_t)PRE_NMS * 4 * 4);
  float* live           = (float*)(ws + off);          off = align256(off + (size_t)PRE_NMS * 4);
  float* areas          = (float*)(ws + off);          off = align256(off + (size_t)PRE_NMS * 4);

  // 1) cast feature map; pack weights into fragment-swizzled bf16
  {
    int n = CIN * NPOS;
    k_cast_bf16<<<(n + 255) / 256, 256, 0, stream>>>(feature_map, Xbf, n);
  }
  {
    int n = COUT * KDIM;
    k_pack_w1<<<(n + 255) / 256, 256, 0, stream>>>(conv1_w, W1swz);
  }
  k_pack_w2<<<(N2 * CIN + 255) / 256, 256, 0, stream>>>(cls_w, box_w, cls_b, box_b, W2swz, bias2);

  // 2) conv1 GEMM (WMMA): 625 M-tiles, 8 waves x 4 N-tiles per block
  k_conv1_wmma<<<NPOS / 16, 256, 0, stream>>>(Xbf, W1swz, conv1_b, Xswz);

  // 3) cls/box GEMM (WMMA): 625*3 waves
  {
    int waves = (NPOS / 16) * (N2 / 16);
    k_gemm2_wmma<<<(waves * 32 + 255) / 256, 256, 0, stream>>>(Xswz, W2swz, bias2, out);
  }

  // 4) decode + clip + masked-score sort keys
  k_decode<<<NSORT / 256, 256, 0, stream>>>(anchor_map, out, proposals, keys);

  // 5) bitonic sort (descending) of 131072 keys
  for (int k = 2; k <= NSORT; k <<= 1)
    for (int j = k >> 1; j > 0; j >>= 1)
      k_bitonic<<<NSORT / 256, 256, 0, stream>>>(keys, k, j);

  // 6) gather top-12000, 7) greedy NMS -> post proposals
  k_prep_nms<<<(PRE_NMS + 255) / 256, 256, 0, stream>>>(keys, proposals, pre_boxes, live, areas);
  k_nms<<<1, 1024, 0, stream>>>(pre_boxes, areas, live, out + NANCH + (size_t)NANCH * 4);
}
